// Explainer_1846835938181
// MI455X (gfx1250) — matrix operand order repaired and checked by reading, not compiled, not run on hardware
//
#include <hip/hip_runtime.h>
#include <math.h>

// ---------------------------------------------------------------------------
// MI455X / gfx1250: split-bf16 WMMA (16x16x32) cdist GEMM with async LDS fills.
// dot(a,b) ~= ahi.bhi + ahi.blo + alo.bhi  (bf16 hi/lo decomposition of f32)
// ---------------------------------------------------------------------------

typedef __bf16 v8bf  __attribute__((ext_vector_type(8)));
typedef __bf16 v16bf __attribute__((ext_vector_type(16)));
typedef float  v8f   __attribute__((ext_vector_type(8)));
typedef int    v4i   __attribute__((ext_vector_type(4)));

#define D_DIM 256
#define NSEG  64

// stage buffer layout (bf16 elements), row stride 40 (32 data + 8 pad)
#define A_ROWS 128
#define B_ROWS 64
#define RSTR   40
#define AHI_OFF 0
#define ALO_OFF (A_ROWS * RSTR)              // 5120
#define BHI_OFF (2 * A_ROWS * RSTR)          // 10240
#define BLO_OFF (2 * A_ROWS * RSTR + B_ROWS * RSTR) // 12800
#define STAGE_BYTES ((2 * A_ROWS + 2 * B_ROWS) * RSTR * 2)  // 30720
#define TAIL_OFF (2 * STAGE_BYTES)           // 61440
#define SMEM_BYTES (TAIL_OFF + 1536)         // 62976

#if defined(__has_builtin)
#if __has_builtin(__builtin_amdgcn_global_load_async_to_lds_b128)
#define USE_ASYNC_LDS 1
#endif
#endif

__device__ __forceinline__ void cp16(const void* gsrc, void* ldst) {
#if defined(USE_ASYNC_LDS)
    __builtin_amdgcn_global_load_async_to_lds_b128(
        (__attribute__((address_space(1))) v4i*)gsrc,
        (__attribute__((address_space(3))) v4i*)ldst, 0, 0);
#else
    *(v4i*)ldst = *(const v4i*)gsrc;
#endif
}
__device__ __forceinline__ void async_wait0() {
#if defined(USE_ASYNC_LDS)
#if __has_builtin(__builtin_amdgcn_s_wait_asynccnt)
    __builtin_amdgcn_s_wait_asynccnt(0);
#else
    asm volatile("s_wait_asynccnt 0" ::: "memory");
#endif
#endif
}

// Order-preserving float <-> uint encoding so atomicMax(u32) == float max.
__device__ __forceinline__ unsigned encf(float f) {
    unsigned u = __float_as_uint(f);
    return (u & 0x80000000u) ? ~u : (u | 0x80000000u);
}
__device__ __forceinline__ float decf(unsigned e) {
    unsigned u = (e & 0x80000000u) ? (e ^ 0x80000000u) : ~e;
    return __uint_as_float(u);
}
#define ENC_NEG_INF 0x007fffffu   // encf(-inf)

__device__ __forceinline__ v16bf mk16(const __bf16* p0, const __bf16* p1) {
    v8bf a = *(const v8bf*)p0;
    v8bf b = *(const v8bf*)p1;
    return __builtin_shufflevector(a, b, 0, 1, 2, 3, 4, 5, 6, 7,
                                   8, 9, 10, 11, 12, 13, 14, 15);
}

// ---------------------------------------------------------------------------
// Kernel 1: ef = 0.5*(h[e0]+h[e1]); emit bf16 hi/lo split + f32 squared norm.
// One block (256 threads == D) per edge.
// ---------------------------------------------------------------------------
__global__ void build_ef_kernel(const float* __restrict__ h,
                                const int* __restrict__ node_edge,
                                const int* __restrict__ label_edge,
                                int En, int El,
                                __bf16* __restrict__ efn_hi, __bf16* __restrict__ efn_lo,
                                __bf16* __restrict__ efl_hi, __bf16* __restrict__ efl_lo,
                                float* __restrict__ nn, float* __restrict__ nl) {
    int e = blockIdx.x;
    int t = threadIdx.x;
    const int* edge; __bf16* ph; __bf16* pl; float* nrm; int idx; int E;
    if (e < En) { edge = node_edge;  ph = efn_hi; pl = efn_lo; nrm = nn; idx = e;      E = En; }
    else        { edge = label_edge; ph = efl_hi; pl = efl_lo; nrm = nl; idx = e - En; E = El; }
    int i0 = edge[idx];
    int i1 = edge[E + idx];            // edge is [2, E] row-major
    float v = 0.5f * (h[(size_t)i0 * D_DIM + t] + h[(size_t)i1 * D_DIM + t]);
    __bf16 hi = (__bf16)v;
    __bf16 lo = (__bf16)(v - (float)hi);
    ph[(size_t)idx * D_DIM + t] = hi;
    pl[(size_t)idx * D_DIM + t] = lo;

    __shared__ float red[D_DIM];
    red[t] = v * v;
    __syncthreads();
    for (int s = D_DIM / 2; s > 0; s >>= 1) {
        if (t < s) red[t] += red[t + s];
        __syncthreads();
    }
    if (t == 0) nrm[idx] = red[0];
}

// ---------------------------------------------------------------------------
// Kernel 2: init encoded-max accumulators to enc(-inf).
// ---------------------------------------------------------------------------
__global__ void init_max_kernel(unsigned* __restrict__ rowmax,
                                unsigned* __restrict__ colmax,
                                int nRow, int nCol) {
    int total = nRow + nCol;
    for (int i = blockIdx.x * blockDim.x + threadIdx.x; i < total;
         i += gridDim.x * blockDim.x) {
        if (i < nRow) rowmax[i] = ENC_NEG_INF;
        else          colmax[i - nRow] = ENC_NEG_INF;
    }
}

// ---------------------------------------------------------------------------
// Kernel 3: split-bf16 WMMA GEMM -> M = -sqrt(max(nr+nc-2dot,0)) with fused
// per-segment max reductions. Block tile 128x64, 8 waves, wave tile 32x32.
// Double-buffered async global->LDS staging of K=32 chunks.
// ---------------------------------------------------------------------------
__global__ void dist_wmma_kernel(const __bf16* __restrict__ efn_hi,
                                 const __bf16* __restrict__ efn_lo,
                                 const __bf16* __restrict__ efl_hi,
                                 const __bf16* __restrict__ efl_lo,
                                 const float* __restrict__ nn,
                                 const float* __restrict__ nl,
                                 const int* __restrict__ node_batch,
                                 const int* __restrict__ label_batch,
                                 unsigned* __restrict__ rowmax,   // [En, NSEG]
                                 unsigned* __restrict__ colmax,   // [NSEG, El]
                                 int En, int El) {
    extern __shared__ char smemraw[];
    float* Mt   = (float*)smemraw;                 // 128 x 65 f32, reuses stages
    float* sN   = (float*)(smemraw + TAIL_OFF);          // 128
    float* sL   = (float*)(smemraw + TAIL_OFF + 512);    // 64
    int*   nbtS = (int*)(smemraw + TAIL_OFF + 768);      // 128
    int*   lblS = (int*)(smemraw + TAIL_OFF + 1280);     // 64

    const int tid  = threadIdx.x;
    const int lane = tid & 31;
    const int wave = tid >> 5;
    const int lr   = lane & 15;
    const int half = lane >> 4;
    const int wm   = wave & 3;    // 0..3 : 32-row group
    const int wn   = wave >> 2;   // 0..1 : 32-col group

    const int r0 = blockIdx.x * 128;
    const int c0 = blockIdx.y * 64;

    // Stage norms + segment ids for this tile.
    if (tid < 128) {
        int r = r0 + tid;
        sN[tid]   = (r < En) ? nn[r] : 0.f;
        nbtS[tid] = (r < En) ? node_batch[r] : -1;
    } else if (tid < 192) {
        int cl = tid - 128;
        int c  = c0 + cl;
        sL[cl]   = (c < El) ? nl[c] : 0.f;
        lblS[cl] = (c < El) ? label_batch[c] : -1;
    }

    v8f acc[2][2];
    #pragma unroll
    for (int mt = 0; mt < 2; ++mt)
        #pragma unroll
        for (int nt = 0; nt < 2; ++nt)
            #pragma unroll
            for (int v = 0; v < 8; ++v) acc[mt][nt][v] = 0.f;

    // stage one K=32 chunk into buffer buf (16B granules, per-lane addresses)
    auto stage = [&](int buf, int kc) {
        __bf16* S = (__bf16*)(smemraw + buf * STAGE_BYTES);
        for (int s = tid; s < A_ROWS * 4; s += 256) {
            int row = s >> 2, seg = s & 3;
            int r = r0 + row; if (r >= En) r = En - 1;
            int lo = row * RSTR + seg * 8;
            cp16(&efn_hi[(size_t)r * D_DIM + kc + seg * 8], &S[AHI_OFF + lo]);
            cp16(&efn_lo[(size_t)r * D_DIM + kc + seg * 8], &S[ALO_OFF + lo]);
        }
        for (int s = tid; s < B_ROWS * 4; s += 256) {
            int row = s >> 2, seg = s & 3;
            int c = c0 + row; if (c >= El) c = El - 1;
            int lo = row * RSTR + seg * 8;
            cp16(&efl_hi[(size_t)c * D_DIM + kc + seg * 8], &S[BHI_OFF + lo]);
            cp16(&efl_lo[(size_t)c * D_DIM + kc + seg * 8], &S[BLO_OFF + lo]);
        }
    };

    const int nchunks = D_DIM / 32;      // 8
    stage(0, 0);
    for (int c = 0; c < nchunks; ++c) {
        async_wait0();
        __syncthreads();
        if (c + 1 < nchunks) stage((c + 1) & 1, (c + 1) * 32);   // overlap

        const __bf16* S = (const __bf16*)(smemraw + (c & 1) * STAGE_BYTES);
        // A frag (16x32 bf16): lane<16 holds K 0-7 & 16-23; lane>=16 K 8-15 & 24-31.
        v16bf ahi[2], alo[2], bhi[2], blo[2];
        #pragma unroll
        for (int mt = 0; mt < 2; ++mt) {
            int ro = (wm * 32 + mt * 16 + lr) * RSTR;
            ahi[mt] = mk16(&S[AHI_OFF + ro + half * 8], &S[AHI_OFF + ro + 16 + half * 8]);
            alo[mt] = mk16(&S[ALO_OFF + ro + half * 8], &S[ALO_OFF + ro + 16 + half * 8]);
        }
        // B frag (32x16): lanes 0-15 -> K 0-7 & 16-23, lanes 16-31 -> K 8-15 & 24-31 for col lr.
        #pragma unroll
        for (int nt = 0; nt < 2; ++nt) {
            int ro = (wn * 32 + nt * 16 + lr) * RSTR + half * 8;
            bhi[nt] = mk16(&S[BHI_OFF + ro], &S[BHI_OFF + ro + 16]);
            blo[nt] = mk16(&S[BLO_OFF + ro], &S[BLO_OFF + ro + 16]);
        }
        // Three bf16 products, interleaved across the 4 accumulators.
        #pragma unroll
        for (int mt = 0; mt < 2; ++mt)
            #pragma unroll
            for (int nt = 0; nt < 2; ++nt)
                acc[mt][nt] = __builtin_amdgcn_wmma_f32_16x16x32_bf16(
                    false, ahi[mt], false, bhi[nt], (short)0, acc[mt][nt], false, false);
        #pragma unroll
        for (int mt = 0; mt < 2; ++mt)
            #pragma unroll
            for (int nt = 0; nt < 2; ++nt)
                acc[mt][nt] = __builtin_amdgcn_wmma_f32_16x16x32_bf16(
                    false, ahi[mt], false, blo[nt], (short)0, acc[mt][nt], false, false);
        #pragma unroll
        for (int mt = 0; mt < 2; ++mt)
            #pragma unroll
            for (int nt = 0; nt < 2; ++nt)
                acc[mt][nt] = __builtin_amdgcn_wmma_f32_16x16x32_bf16(
                    false, alo[mt], false, bhi[nt], (short)0, acc[mt][nt], false, false);
        __syncthreads();
    }

    // dot -> M = -sqrt(max(nr+nc-2dot,0)) into LDS tile.
    // C/D layout: VGPR v: lanes 0-15 -> M=v, lanes 16-31 -> M=v+8; N=lane&15.
    #pragma unroll
    for (int mt = 0; mt < 2; ++mt)
        #pragma unroll
        for (int nt = 0; nt < 2; ++nt)
            #pragma unroll
            for (int v = 0; v < 8; ++v) {
                int lrow = wm * 32 + mt * 16 + half * 8 + v;
                int lcol = wn * 32 + nt * 16 + lr;
                float dot = acc[mt][nt][v];
                float sq  = sN[lrow] + sL[lcol] - 2.f * dot;
                Mt[lrow * 65 + lcol] = (sq > 0.f) ? -sqrtf(sq) : 0.f;
            }
    __syncthreads();

    // Fused segment reductions (batch arrays are sorted -> runs are contiguous).
    if (tid < 128) {
        int r = r0 + tid;
        if (r < En) {
            float best = -INFINITY;
            int cur = -1;
            for (int j = 0; j < 64; ++j) {
                int seg = lblS[j];
                if (seg < 0) break;
                if (seg != cur) {
                    if (cur >= 0)
                        atomicMax(&rowmax[(size_t)r * NSEG + cur], encf(best));
                    cur = seg;
                    best = -INFINITY;
                }
                best = fmaxf(best, Mt[tid * 65 + j]);
            }
            if (cur >= 0)
                atomicMax(&rowmax[(size_t)r * NSEG + cur], encf(best));
        }
    } else if (tid < 192) {
        int cl = tid - 128;
        int c  = c0 + cl;
        if (c < El) {
            float best = -INFINITY;
            int cur = -1;
            for (int j = 0; j < 128; ++j) {
                int seg = nbtS[j];
                if (seg < 0) break;
                if (seg != cur) {
                    if (cur >= 0)
                        atomicMax(&colmax[(size_t)cur * El + c], encf(best));
                    cur = seg;
                    best = -INFINITY;
                }
                best = fmaxf(best, Mt[j * 65 + cl]);
            }
            if (cur >= 0)
                atomicMax(&colmax[(size_t)cur * El + c], encf(best));
        }
    }
}

// ---------------------------------------------------------------------------
// Kernel 4: segment means + combine. One thread per (gn, gl).
// ---------------------------------------------------------------------------
__device__ __forceinline__ int lowerb(const int* a, int n, int v) {
    int lo = 0, hi = n;
    while (lo < hi) { int m = (lo + hi) >> 1; if (a[m] <  v) lo = m + 1; else hi = m; }
    return lo;
}
__device__ __forceinline__ int upperb(const int* a, int n, int v) {
    int lo = 0, hi = n;
    while (lo < hi) { int m = (lo + hi) >> 1; if (a[m] <= v) lo = m + 1; else hi = m; }
    return lo;
}

__global__ void finalize_kernel(const unsigned* __restrict__ rowmax,
                                const unsigned* __restrict__ colmax,
                                const int* __restrict__ node_batch,
                                const int* __restrict__ label_batch,
                                int En, int El, float* __restrict__ out) {
    int idx = blockIdx.x * blockDim.x + threadIdx.x;
    if (idx >= NSEG * NSEG) return;
    int gn = idx >> 6, gl = idx & 63;

    int ns = lowerb(node_batch, En, gn),  ne = upperb(node_batch, En, gn);
    int ls = lowerb(label_batch, El, gl), le = upperb(label_batch, El, gl);

    float sn = 0.f;
    for (int r = ns; r < ne; ++r) {
        float v = decf(rowmax[(size_t)r * NSEG + gl]);
        if (v == -INFINITY) v = 0.f;   // empty label segment -> 0 (isfinite rule)
        sn += v;
    }
    float on = (ne > ns) ? sn / (float)(ne - ns) : 0.f;

    float sl = 0.f;
    for (int c = ls; c < le; ++c) {
        float v = decf(colmax[(size_t)gn * El + c]);
        if (v == -INFINITY) v = 0.f;   // empty node segment -> 0
        sl += v;
    }
    float ol = (le > ls) ? sl / (float)(le - ls) : 0.f;

    out[idx] = 0.5f * (on + ol);
}

// ---------------------------------------------------------------------------
extern "C" void kernel_launch(void* const* d_in, const int* in_sizes, int n_in,
                              void* d_out, int out_size, void* d_ws, size_t ws_size,
                              hipStream_t stream) {
    (void)n_in; (void)out_size; (void)ws_size;
    const float* h           = (const float*)d_in[0];
    const int*   node_edge   = (const int*)d_in[1];
    const int*   node_batch  = (const int*)d_in[2];
    const int*   label_edge  = (const int*)d_in[3];
    const int*   label_batch = (const int*)d_in[4];
    const int En = in_sizes[1] / 2;
    const int El = in_sizes[3] / 2;

    char* ws = (char*)d_ws;
    auto align256 = [](size_t x) { return (x + 255) & ~(size_t)255; };
    size_t off = 0;
    __bf16* efn_hi = (__bf16*)(ws + off); off = align256(off + (size_t)En * D_DIM * 2);
    __bf16* efn_lo = (__bf16*)(ws + off); off = align256(off + (size_t)En * D_DIM * 2);
    __bf16* efl_hi = (__bf16*)(ws + off); off = align256(off + (size_t)El * D_DIM * 2);
    __bf16* efl_lo = (__bf16*)(ws + off); off = align256(off + (size_t)El * D_DIM * 2);
    float* nn = (float*)(ws + off); off = align256(off + (size_t)En * sizeof(float));
    float* nl = (float*)(ws + off); off = align256(off + (size_t)El * sizeof(float));
    unsigned* rowmax = (unsigned*)(ws + off); off = align256(off + (size_t)En * NSEG * 4);
    unsigned* colmax = (unsigned*)(ws + off); off = align256(off + (size_t)NSEG * El * 4);

    build_ef_kernel<<<En + El, D_DIM, 0, stream>>>(h, node_edge, label_edge, En, El,
                                                   efn_hi, efn_lo, efl_hi, efl_lo, nn, nl);

    int initTotal = En * NSEG + NSEG * El;
    init_max_kernel<<<(initTotal + 255) / 256, 256, 0, stream>>>(rowmax, colmax,
                                                                 En * NSEG, NSEG * El);

    dim3 grid((En + 127) / 128, (El + 63) / 64);
    dist_wmma_kernel<<<grid, 256, SMEM_BYTES, stream>>>(
        efn_hi, efn_lo, efl_hi, efl_lo, nn, nl,
        node_batch, label_batch, rowmax, colmax, En, El);

    finalize_kernel<<<(NSEG * NSEG + 255) / 256, 256, 0, stream>>>(
        rowmax, colmax, node_batch, label_batch, En, El, (float*)d_out);
}
